// Block_53979148976597
// MI455X (gfx1250) — matrix operand order repaired and checked by reading, not compiled, not run on hardware
//
#include <hip/hip_runtime.h>
#include <hip/hip_bf16.h>

// ---------------------------------------------------------------------------
// BEiT-style transformer block for MI455X (gfx1250, wave32, WMMA f16->f32).
// All matmuls run through v_wmma_f32_16x16x32_f16. GEMM B-panels are staged
// through LDS with double-buffered GLOBAL_LOAD_ASYNC_TO_LDS_B128 (ASYNCcnt),
// steady-state loop peeled so the hot body is branch-free:
//   A b128 loads -> async stage(k+1) -> s_wait_asynccnt 8 -> ds frags -> WMMA.
// Attention is flash-style with LDS P-transpose and explicit s_wait_dscnt.
// ---------------------------------------------------------------------------

typedef _Float16 half_t;
typedef __attribute__((ext_vector_type(16))) _Float16 v16h;
typedef __attribute__((ext_vector_type(8)))  _Float16 v8h;
typedef __attribute__((ext_vector_type(8)))  float    v8f;

union HV { v16h v; v8h h[2]; };

static __device__ __forceinline__ v8f vzero8f() {
  v8f z = {0.f, 0.f, 0.f, 0.f, 0.f, 0.f, 0.f, 0.f};
  return z;
}

#define WMMA_F16(a, b, c) \
  __builtin_amdgcn_wmma_f32_16x16x32_f16(false, (a), false, (b), (short)0, (c), false, false)

// ---------------------------------------------------------------------------
// Wave-level GEMM tile: acc[4] (16x64 f32) += A[16xK] * B[NxK]^T.
// A row-major [M,K] f16 (direct global b128 fragment loads).
// B row-major [N,K] f16, staged per-wave into LDS with double-buffered
// async global->LDS copies. Last K-step peeled so the loop body needs no
// branches around the ASYNCcnt waits.
// bs points at this wave's 2*64*32-half LDS staging area.
// ---------------------------------------------------------------------------
static __device__ __forceinline__ void gemm16x64(const half_t* __restrict__ Ah,
                                                 const half_t* __restrict__ Bh,
                                                 half_t* bs,
                                                 int rowBase, int colBase, int K,
                                                 v8f acc[4]) {
  const int lane = threadIdx.x & 31;
  const int mrow = lane & 15;
  const int kbA  = (lane & 16) ? 8  : 0;
  const int kbB  = (lane & 16) ? 16 : 0;
  const half_t* arow = Ah + (size_t)(rowBase + mrow) * K + kbA;

  // async-stage one 64x32 B panel (4KB) into LDS buffer `buf`
  const int srow = lane >> 2;         // 0..7: panel row within group of 8
  const int scol = (lane & 3) * 8;    // 16B chunk within the 32-half row
  auto stage = [&](int buf, int k0) {
    #pragma unroll
    for (int j = 0; j < 8; ++j) {
      const int rp = j * 8 + srow;
      const half_t* g = Bh + (size_t)(colBase + rp) * K + k0 + scol;
      unsigned lds = (unsigned)(uintptr_t)(bs + buf * 2048 + rp * 32 + scol);
      asm volatile("global_load_async_to_lds_b128 %0, %1, off"
                   :: "v"(lds), "v"(g) : "memory");
    }
  };

  auto compute = [&](int buf, const HV& a) {
    const half_t* bp = bs + buf * 2048 + mrow * 32 + kbA;
    HV b0, b1, b2, b3;
    b0.h[0] = *(const v8h*)(bp +    0); b0.h[1] = *(const v8h*)(bp +   16);
    b1.h[0] = *(const v8h*)(bp +  512); b1.h[1] = *(const v8h*)(bp +  528);
    b2.h[0] = *(const v8h*)(bp + 1024); b2.h[1] = *(const v8h*)(bp + 1040);
    b3.h[0] = *(const v8h*)(bp + 1536); b3.h[1] = *(const v8h*)(bp + 1552);
    acc[0] = WMMA_F16(a.v, b0.v, acc[0]);
    acc[1] = WMMA_F16(a.v, b1.v, acc[1]);
    acc[2] = WMMA_F16(a.v, b2.v, acc[2]);
    acc[3] = WMMA_F16(a.v, b3.v, acc[3]);
  };

  stage(0, 0);
  int k0 = 0;
  // steady state: branch-free body, one panel always in flight
  for (; k0 + 32 < K; k0 += 32) {
    __builtin_prefetch(arow + k0 + 128, 0, 0);   // global_prefetch_b8
    HV a;
    a.h[0] = *(const v8h*)(arow + k0);
    a.h[1] = *(const v8h*)(arow + k0 + 16);
    const int buf = (k0 >> 5) & 1;
    stage(buf ^ 1, k0 + 32);                     // prefetch next panel
    asm volatile("s_wait_asynccnt 0x8" ::: "memory");  // stage k complete
    compute(buf, a);
  }
  // peeled final step: drain ASYNCcnt fully
  {
    HV a;
    a.h[0] = *(const v8h*)(arow + k0);
    a.h[1] = *(const v8h*)(arow + k0 + 16);
    asm volatile("s_wait_asynccnt 0x0" ::: "memory");
    compute((k0 >> 5) & 1, a);
  }
}

// ---------------------------------------------------------------------------
// fp32 -> f16 weight conversion
// ---------------------------------------------------------------------------
__global__ void f32_to_f16_kernel(const float* __restrict__ src,
                                  half_t* __restrict__ dst, int n) {
  int i = blockIdx.x * 256 + threadIdx.x;
  if (i < n) dst[i] = (half_t)src[i];
}

// ---------------------------------------------------------------------------
// LayerNorm over 768 channels, one block (8 waves) per row; f16 output.
// ---------------------------------------------------------------------------
__global__ void ln_kernel(const float* __restrict__ src,
                          const float* __restrict__ w,
                          const float* __restrict__ bia,
                          half_t* __restrict__ dst) {
  const int row = blockIdx.x;
  const float* xr = src + (size_t)row * 768;
  float s = 0.f, sq = 0.f;
  for (int i = threadIdx.x; i < 768; i += 256) {
    float v = xr[i];
    s += v; sq += v * v;
  }
  #pragma unroll
  for (int m = 1; m < 32; m <<= 1) {
    s  += __shfl_xor(s,  m);
    sq += __shfl_xor(sq, m);
  }
  __shared__ float rs[8], rq[8];
  __shared__ float mean_s, rstd_s;
  const int wid = threadIdx.x >> 5;
  if ((threadIdx.x & 31) == 0) { rs[wid] = s; rq[wid] = sq; }
  __syncthreads();
  if (threadIdx.x == 0) {
    float S = 0.f, Q = 0.f;
    #pragma unroll
    for (int i = 0; i < 8; ++i) { S += rs[i]; Q += rq[i]; }
    float mean = S * (1.0f / 768.0f);
    float var  = Q * (1.0f / 768.0f) - mean * mean;
    mean_s = mean;
    rstd_s = rsqrtf(var + 1e-5f);
  }
  __syncthreads();
  const float mean = mean_s, rstd = rstd_s;
  for (int i = threadIdx.x; i < 768; i += 256)
    dst[(size_t)row * 768 + i] = (half_t)((xr[i] - mean) * rstd * w[i] + bia[i]);
}

// ---------------------------------------------------------------------------
// QKV GEMM: t[8192,768] @ qkv_w[2304,768]^T ; epilogue splits q/k/v.
// q scaled by 1/8 and biased; k unbiased; v biased and stored TRANSPOSED
// as vt[bh][d][n] so attention B-fragments are contiguous.
// ---------------------------------------------------------------------------
__global__ void __launch_bounds__(128)
qkv_gemm_kernel(const half_t* __restrict__ Ah, const half_t* __restrict__ Bh,
                const float* __restrict__ q_bias, const float* __restrict__ v_bias,
                half_t* __restrict__ qh, half_t* __restrict__ kh,
                half_t* __restrict__ vt) {
  __shared__ __align__(16) half_t Bstage[4][2 * 64 * 32];
  const int wid  = threadIdx.x >> 5;
  const int wave = (blockIdx.x * blockDim.x + threadIdx.x) >> 5;
  const int CG = 2304 / 64;
  const int rowBase = (wave / CG) * 16;
  const int colBase = (wave % CG) * 64;
  v8f acc[4];
  #pragma unroll
  for (int t = 0; t < 4; ++t) acc[t] = vzero8f();
  gemm16x64(Ah, Bh, &Bstage[wid][0], rowBase, colBase, 768, acc);

  const int lane = threadIdx.x & 31;
  const int mrow = lane & 15;
  const int hi   = (lane >> 4) & 1;
  const int sec  = colBase / 768;      // 0=q 1=k 2=v
  const int rem  = colBase % 768;      // head*64
  const int head = rem >> 6;
  #pragma unroll
  for (int t = 0; t < 4; ++t) {
    const int d = t * 16 + mrow;
    #pragma unroll
    for (int r = 0; r < 8; ++r) {
      const int m = rowBase + r + 8 * hi;
      const int b = m >> 10, ntok = m & 1023;
      const int bh = b * 12 + head;
      float v = acc[t][r];
      if (sec == 0) {
        v = (v + q_bias[rem + d]) * 0.125f;  // HD^-0.5 = 1/8
        qh[((size_t)bh * 1024 + ntok) * 64 + d] = (half_t)v;
      } else if (sec == 1) {
        kh[((size_t)bh * 1024 + ntok) * 64 + d] = (half_t)v;
      } else {
        v += v_bias[rem + d];
        vt[((size_t)bh * 64 + d) * 1024 + ntok] = (half_t)v;
      }
    }
  }
}

// ---------------------------------------------------------------------------
// Flash attention: one wave = 16 query rows x full head dim (64).
// Streams 32-key tiles: S = q@k^T (4 WMMAs) + rel-pos-bias gather,
// online softmax (shfl_xor row reductions inside 16-lane halves),
// P staged via LDS (C-layout -> A-layout), acc += P@V (4 WMMAs).
// ---------------------------------------------------------------------------
__global__ void __launch_bounds__(128)
attn_kernel(const half_t* __restrict__ qh, const half_t* __restrict__ kh,
            const half_t* __restrict__ vt,
            const float* __restrict__ rel_table, const int* __restrict__ rel_idx,
            half_t* __restrict__ ao) {
  __shared__ __align__(16) half_t Pbuf[4][16][32];
  const int wid  = threadIdx.x >> 5;
  const int lane = threadIdx.x & 31;
  const int task = blockIdx.x * 4 + wid;        // 96 bh * 64 row tiles
  const int rowTile = task & 63;
  const int bh   = task >> 6;
  const int head = bh % 12;
  const int b    = bh / 12;
  const int rowBase = rowTile << 4;
  const half_t* qp = qh + (size_t)bh * (1024 * 64);
  const half_t* kp = kh + (size_t)bh * (1024 * 64);
  const half_t* vp = vt + (size_t)bh * (64 * 1024);
  const int mrow = lane & 15;
  const int hi   = (lane >> 4) & 1;
  const int kbA  = hi ? 8 : 0;
  const int kbB  = hi ? 16 : 0;

  HV qa0, qa1;  // A-fragments for K-dim 0..31 and 32..63
  {
    const half_t* qrow = qp + (size_t)(rowBase + mrow) * 64;
    qa0.h[0] = *(const v8h*)(qrow + kbA);
    qa0.h[1] = *(const v8h*)(qrow + 16 + kbA);
    qa1.h[0] = *(const v8h*)(qrow + 32 + kbA);
    qa1.h[1] = *(const v8h*)(qrow + 48 + kbA);
  }

  float rowM[8], rowL[8];
  v8f acc[4];
  #pragma unroll
  for (int r = 0; r < 8; ++r) { rowM[r] = -3.0e38f; rowL[r] = 0.f; }
  #pragma unroll
  for (int t = 0; t < 4; ++t) acc[t] = vzero8f();

  for (int mb = 0; mb < 1024; mb += 32) {
    v8f s[2];
    s[0] = vzero8f(); s[1] = vzero8f();
    #pragma unroll
    for (int ct = 0; ct < 2; ++ct) {
      const half_t* krow = kp + (size_t)(mb + ct * 16 + mrow) * 64;
      HV b0, b1;
      b0.h[0] = *(const v8h*)(krow + kbA);
      b0.h[1] = *(const v8h*)(krow + kbA + 16);
      b1.h[0] = *(const v8h*)(krow + 32 + kbA);
      b1.h[1] = *(const v8h*)(krow + 32 + kbA + 16);
      s[ct] = WMMA_F16(qa0.v, b0.v, s[ct]);
      s[ct] = WMMA_F16(qa1.v, b1.v, s[ct]);
    }
    // relative position bias gather: rpb[head][n][m]
    #pragma unroll
    for (int ct = 0; ct < 2; ++ct) {
      const int mcol = mb + ct * 16 + mrow;
      #pragma unroll
      for (int r = 0; r < 8; ++r) {
        const int ntok = rowBase + r + 8 * hi;
        const int idx  = rel_idx[(size_t)ntok * 1024 + mcol];
        s[ct][r] += rel_table[(size_t)idx * 12 + head];
      }
    }
    // online softmax; each 16-lane half shares its 8 rows.
    #pragma unroll
    for (int r = 0; r < 8; ++r) {
      float mx = fmaxf(s[0][r], s[1][r]);
      mx = fmaxf(mx, __shfl_xor(mx, 1));
      mx = fmaxf(mx, __shfl_xor(mx, 2));
      mx = fmaxf(mx, __shfl_xor(mx, 4));
      mx = fmaxf(mx, __shfl_xor(mx, 8));
      const float newM  = fmaxf(rowM[r], mx);
      const float scale = __expf(rowM[r] - newM);
      const float p0 = __expf(s[0][r] - newM);
      const float p1 = __expf(s[1][r] - newM);
      float sum = p0 + p1;
      sum += __shfl_xor(sum, 1);
      sum += __shfl_xor(sum, 2);
      sum += __shfl_xor(sum, 4);
      sum += __shfl_xor(sum, 8);
      rowL[r] = rowL[r] * scale + sum;
      rowM[r] = newM;
      #pragma unroll
      for (int dt = 0; dt < 4; ++dt) acc[dt][r] *= scale;
      Pbuf[wid][r + 8 * hi][mrow]      = (half_t)p0;
      Pbuf[wid][r + 8 * hi][mrow + 16] = (half_t)p1;
    }
    // DS ops are in-order per wave; drain before cross-lane A-layout reload.
    asm volatile("s_wait_dscnt 0x0" ::: "memory");
    HV pa;
    {
      const half_t* prow = &Pbuf[wid][mrow][0];
      pa.h[0] = *(const v8h*)(prow + kbA);
      pa.h[1] = *(const v8h*)(prow + 16 + kbA);
    }
    #pragma unroll
    for (int dt = 0; dt < 4; ++dt) {
      const half_t* vrow = vp + (size_t)(dt * 16 + mrow) * 1024 + mb + kbA;
      HV vb;
      vb.h[0] = *(const v8h*)(vrow);
      vb.h[1] = *(const v8h*)(vrow + 16);
      acc[dt] = WMMA_F16(pa.v, vb.v, acc[dt]);
    }
    asm volatile("s_wait_dscnt 0x0" ::: "memory");
  }
  #pragma unroll
  for (int r = 0; r < 8; ++r) {
    const float inv = 1.0f / rowL[r];
    const int ntok = rowBase + r + 8 * hi;
    const size_t base = ((size_t)(b * 1024 + ntok)) * 768 + head * 64;
    #pragma unroll
    for (int dt = 0; dt < 4; ++dt)
      ao[base + dt * 16 + mrow] = (half_t)(acc[dt][r] * inv);
  }
}

// ---------------------------------------------------------------------------
// proj GEMM + residual: x1 = x + gamma1 * (ao @ proj_w^T + proj_b)
// ---------------------------------------------------------------------------
__global__ void __launch_bounds__(128)
proj_gemm_kernel(const half_t* __restrict__ Ah, const half_t* __restrict__ Bh,
                 const float* __restrict__ bias, const float* __restrict__ xin,
                 const float* __restrict__ gamma, float* __restrict__ x1) {
  __shared__ __align__(16) half_t Bstage[4][2 * 64 * 32];
  const int wid  = threadIdx.x >> 5;
  const int wave = (blockIdx.x * blockDim.x + threadIdx.x) >> 5;
  const int CG = 768 / 64;
  const int rowBase = (wave / CG) * 16;
  const int colBase = (wave % CG) * 64;
  v8f acc[4];
  #pragma unroll
  for (int t = 0; t < 4; ++t) acc[t] = vzero8f();
  gemm16x64(Ah, Bh, &Bstage[wid][0], rowBase, colBase, 768, acc);
  const int lane = threadIdx.x & 31;
  const int mrow = lane & 15;
  const int hi   = (lane >> 4) & 1;
  #pragma unroll
  for (int t = 0; t < 4; ++t) {
    const int n = colBase + t * 16 + mrow;
    #pragma unroll
    for (int r = 0; r < 8; ++r) {
      const int m = rowBase + r + 8 * hi;
      x1[(size_t)m * 768 + n] =
          xin[(size_t)m * 768 + n] + gamma[n] * (acc[t][r] + bias[n]);
    }
  }
}

// ---------------------------------------------------------------------------
// FC1 GEMM + exact GELU -> f16
// ---------------------------------------------------------------------------
__global__ void __launch_bounds__(128)
fc1_gemm_kernel(const half_t* __restrict__ Ah, const half_t* __restrict__ Bh,
                const float* __restrict__ bias, half_t* __restrict__ out) {
  __shared__ __align__(16) half_t Bstage[4][2 * 64 * 32];
  const int wid  = threadIdx.x >> 5;
  const int wave = (blockIdx.x * blockDim.x + threadIdx.x) >> 5;
  const int CG = 3072 / 64;
  const int rowBase = (wave / CG) * 16;
  const int colBase = (wave % CG) * 64;
  v8f acc[4];
  #pragma unroll
  for (int t = 0; t < 4; ++t) acc[t] = vzero8f();
  gemm16x64(Ah, Bh, &Bstage[wid][0], rowBase, colBase, 768, acc);
  const int lane = threadIdx.x & 31;
  const int mrow = lane & 15;
  const int hi   = (lane >> 4) & 1;
  #pragma unroll
  for (int t = 0; t < 4; ++t) {
    const int n = colBase + t * 16 + mrow;
    #pragma unroll
    for (int r = 0; r < 8; ++r) {
      const int m = rowBase + r + 8 * hi;
      const float v = acc[t][r] + bias[n];
      const float g = 0.5f * v * (1.0f + erff(v * 0.70710678118654752f));
      out[(size_t)m * 3072 + n] = (half_t)g;
    }
  }
}

// ---------------------------------------------------------------------------
// FC2 GEMM + residual: out = x1 + gamma2 * (mh @ fc2_w^T + fc2_b)
// ---------------------------------------------------------------------------
__global__ void __launch_bounds__(128)
fc2_gemm_kernel(const half_t* __restrict__ Ah, const half_t* __restrict__ Bh,
                const float* __restrict__ bias, const float* __restrict__ x1,
                const float* __restrict__ gamma, float* __restrict__ out) {
  __shared__ __align__(16) half_t Bstage[4][2 * 64 * 32];
  const int wid  = threadIdx.x >> 5;
  const int wave = (blockIdx.x * blockDim.x + threadIdx.x) >> 5;
  const int CG = 768 / 64;
  const int rowBase = (wave / CG) * 16;
  const int colBase = (wave % CG) * 64;
  v8f acc[4];
  #pragma unroll
  for (int t = 0; t < 4; ++t) acc[t] = vzero8f();
  gemm16x64(Ah, Bh, &Bstage[wid][0], rowBase, colBase, 3072, acc);
  const int lane = threadIdx.x & 31;
  const int mrow = lane & 15;
  const int hi   = (lane >> 4) & 1;
  #pragma unroll
  for (int t = 0; t < 4; ++t) {
    const int n = colBase + t * 16 + mrow;
    #pragma unroll
    for (int r = 0; r < 8; ++r) {
      const int m = rowBase + r + 8 * hi;
      out[(size_t)m * 768 + n] =
          x1[(size_t)m * 768 + n] + gamma[n] * (acc[t][r] + bias[n]);
    }
  }
}

// ---------------------------------------------------------------------------
extern "C" void kernel_launch(void* const* d_in, const int* in_sizes, int n_in,
                              void* d_out, int out_size, void* d_ws, size_t ws_size,
                              hipStream_t stream) {
  (void)in_sizes; (void)n_in; (void)out_size; (void)ws_size;
  const float* x        = (const float*)d_in[0];
  const float* norm1_w  = (const float*)d_in[1];
  const float* norm1_b  = (const float*)d_in[2];
  const float* qkv_w    = (const float*)d_in[3];
  const float* q_bias   = (const float*)d_in[4];
  const float* v_bias   = (const float*)d_in[5];
  const float* proj_w   = (const float*)d_in[6];
  const float* proj_b   = (const float*)d_in[7];
  const float* rel_tab  = (const float*)d_in[8];
  const float* norm2_w  = (const float*)d_in[9];
  const float* norm2_b  = (const float*)d_in[10];
  const float* fc1_w    = (const float*)d_in[11];
  const float* fc1_b    = (const float*)d_in[12];
  const float* fc2_w    = (const float*)d_in[13];
  const float* fc2_b    = (const float*)d_in[14];
  const float* gamma1   = (const float*)d_in[15];
  const float* gamma2   = (const float*)d_in[16];
  const int*   rel_idx  = (const int*)d_in[17];
  float* out = (float*)d_out;

  // workspace carve-up
  char* p = (char*)d_ws;
  auto take = [&](size_t bytes) {
    char* r = p;
    p += (bytes + 255) & ~(size_t)255;
    return r;
  };
  half_t* t_h  = (half_t*)take(8192ull * 768 * 2);   // LN1 output (f16)
  half_t* qh   = (half_t*)take(96ull * 1024 * 64 * 2);
  half_t* kh   = (half_t*)take(96ull * 1024 * 64 * 2);
  half_t* vt   = (half_t*)take(96ull * 64 * 1024 * 2); // V transposed
  half_t* ao_h = (half_t*)take(8192ull * 768 * 2);   // attention out (f16)
  float*  x1   = (float*) take(8192ull * 768 * 4);   // first residual (f32)
  half_t* h2   = (half_t*)take(8192ull * 768 * 2);   // LN2 output (f16)
  half_t* mh   = (half_t*)take(8192ull * 3072 * 2);  // GELU(FC1) (f16)
  half_t* Wq   = (half_t*)take(2304ull * 768 * 2);
  half_t* Wp   = (half_t*)take(768ull * 768 * 2);
  half_t* W1   = (half_t*)take(3072ull * 768 * 2);
  half_t* W2   = (half_t*)take(768ull * 3072 * 2);

  // 1) weights -> f16
  f32_to_f16_kernel<<<(2304 * 768 + 255) / 256, 256, 0, stream>>>(qkv_w, Wq, 2304 * 768);
  f32_to_f16_kernel<<<(768 * 768 + 255) / 256, 256, 0, stream>>>(proj_w, Wp, 768 * 768);
  f32_to_f16_kernel<<<(3072 * 768 + 255) / 256, 256, 0, stream>>>(fc1_w, W1, 3072 * 768);
  f32_to_f16_kernel<<<(768 * 3072 + 255) / 256, 256, 0, stream>>>(fc2_w, W2, 768 * 3072);

  // 2) LN1
  ln_kernel<<<8192, 256, 0, stream>>>(x, norm1_w, norm1_b, t_h);
  // 3) QKV GEMM (8192x768 @ 2304x768^T): 512 row tiles x 36 col groups
  qkv_gemm_kernel<<<(512 * 36) / 4, 128, 0, stream>>>(t_h, Wq, q_bias, v_bias, qh, kh, vt);
  // 4) attention: 96 (b,h) x 64 row tiles
  attn_kernel<<<(96 * 64) / 4, 128, 0, stream>>>(qh, kh, vt, rel_tab, rel_idx, ao_h);
  // 5) proj + residual(gamma1)
  proj_gemm_kernel<<<(512 * 12) / 4, 128, 0, stream>>>(ao_h, Wp, proj_b, x, gamma1, x1);
  // 6) LN2
  ln_kernel<<<8192, 256, 0, stream>>>(x1, norm2_w, norm2_b, h2);
  // 7) FC1 + GELU
  fc1_gemm_kernel<<<(512 * 48) / 4, 128, 0, stream>>>(h2, W1, fc1_b, mh);
  // 8) FC2 + residual(gamma2) -> output
  fc2_gemm_kernel<<<(512 * 12) / 4, 128, 0, stream>>>(mh, W2, fc2_b, x1, gamma2, out);
}
